// AGE_61211873902622
// MI455X (gfx1250) — matrix operand order repaired and checked by reading, not compiled, not run on hardware
//
#include <hip/hip_runtime.h>

// ---------------------------------------------------------------------------
// Fused  out = tanh( norm(tgt) @ norm(tgt)^T ) @ tgt   for [B=8, N=4096, D=512]
//
// MI455X reasoning: tgt (64MB) is L2-resident (192MB L2); stream 32-row chunks
// of tgt from L2, never materialize the BxNxN sim. Both GEMMs run on
// v_wmma_f32_16x16x32_bf16. The chunk is staged in LDS ONCE (bf16, transposed
// [d][m]); GEMM1 reads it as contiguous rows, GEMM2 reads it through
// plain ds_load_b128 of contiguous m in row d. tanh uses the gfx1250
// hardware v_tanh_f32; bf16 packing goes through v_cvt_pk_bf16_f32.
// ---------------------------------------------------------------------------

typedef __attribute__((ext_vector_type(16))) __bf16   v16bf;
typedef __attribute__((ext_vector_type(2)))  __bf16   v2bf;
typedef __attribute__((ext_vector_type(8)))  float    v8f;
typedef __attribute__((ext_vector_type(4)))  unsigned v4u;
typedef __attribute__((ext_vector_type(8)))  unsigned v8u;

constexpr int kB = 8, kN = 4096, kD = 512;
constexpr int ROWS     = 32;   // output rows per workgroup (2 row-tiles of 16)
constexpr int CH       = 32;   // m-chunk size == bf16 WMMA K
constexpr int NTHREADS = 256;  // 8 wave32s

// pair of f32 -> packed bf16 word (backend emits v_cvt_pk_bf16_f32)
static __device__ __forceinline__ unsigned packbf(float a, float b) {
  v2bf p = { (__bf16)a, (__bf16)b };
  return __builtin_bit_cast(unsigned, p);
}

// gfx1250 hardware tanh (TRANS32 op); trailing v_nop covers the
// trans-result hazard (1 independent op before the output is consumed).
static __device__ __forceinline__ float htanh(float x) {
  float y;
  asm volatile("v_tanh_f32 %0, %1\n\tv_nop" : "=v"(y) : "v"(x));
  return y;
}

// ---- phase 1: rinv[row] = 1/||tgt[row,:]||  (one wave32 per row) -----------
__global__ void age_rownorm(const float* __restrict__ tgt, float* __restrict__ rinv) {
  int gw   = (blockIdx.x * blockDim.x + threadIdx.x) >> 5;  // global wave id == row
  int lane = threadIdx.x & 31;
  const float* p = tgt + (size_t)gw * kD;
  float s = 0.f;
  for (int c = lane * 4; c < kD; c += 128) {
    float4 v = *reinterpret_cast<const float4*>(p + c);
    s += v.x * v.x + v.y * v.y + v.z * v.z + v.w * v.w;
  }
  for (int off = 16; off; off >>= 1) s += __shfl_xor(s, off, 32);
  if (lane == 0) rinv[gw] = rsqrtf(s);
}

// ---- phase 2: fused double-GEMM ------------------------------------------
__global__ __launch_bounds__(NTHREADS)
void age_fused(const float* __restrict__ tgt, const float* __restrict__ rinv,
               float* __restrict__ out) {
  __shared__ __bf16 bufT[CH * kD];   // chunk, transposed: bufT[d*CH + m]
  __shared__ float  S32[ROWS * CH];  // f32 sim partials for this row block
  __shared__ float  lrinv[CH];       // rinv of the chunk's m rows

  const int tid  = threadIdx.x;
  const int lane = tid & 31;
  const int wave = tid >> 5;
  const int rt   = wave >> 2;        // row-tile within block (0..1)
  const int q    = wave & 3;         // D-quarter (0..3), 128 columns each

  const int b        = blockIdx.x >> 7;               // kN/ROWS = 128 blocks/batch
  const int blockRow = (blockIdx.x & 127) * ROWS;
  const float* tb = tgt + (size_t)b * kN * kD;

  // --- persistent register A-fragments: normalized rows, WMMA A layout ---
  // lanes 0-15 hold K = {0..7, 16..23}; lanes 16-31 hold K = {8..15, 24..31}
  const int arow  = rt * 16 + (lane & 15);
  const int shalf = (lane & 16) ? 8 : 0;
  const float* ap = tb + (size_t)(blockRow + arow) * kD;
  const float rr  = rinv[b * kN + blockRow + arow];
  v16bf afrag[4];
  for (int t = 0; t < 4; ++t) {
    int k0 = q * 128 + t * 32;
    v8u w;
    for (int j = 0; j < 4; ++j) {
      w[j]     = packbf(ap[k0 + shalf + 2 * j] * rr,      ap[k0 + shalf + 2 * j + 1] * rr);
      w[4 + j] = packbf(ap[k0 + 16 + shalf + 2 * j] * rr, ap[k0 + 16 + shalf + 2 * j + 1] * rr);
    }
    afrag[t] = __builtin_bit_cast(v16bf, w);
  }

  v8f acc[8];
  for (int i = 0; i < 8; ++i) acc[i] = {};

  // staging: thread owns 2 chunk rows (m, m+1) x 32 d-columns so bf16 pairs
  // along m become single b32 LDS stores.
  const int mp    = (tid >> 4) * 2;     // m0 (0,2,..,30)
  const int dbase = (tid & 15) * 32;    // 32 d-columns per thread

  // per-lane byte offset inside a 16(d) x 16(m) bufT tile (rows are CH*2 = 64B apart)
  const unsigned trlane = (unsigned)((lane & 15) * (CH * 2) + ((lane >> 4) << 4));

  for (int mc = 0; mc < kN; mc += CH) {
    __syncthreads();  // prior GEMM2 finished with bufT / S32

    // --- stage chunk once: bf16, transposed [d][m]; zero S32; chunk rinv ---
    const float* src = tb + (size_t)mc * kD;
    const float* r0  = src + (size_t)mp * kD;
    const float* r1  = r0 + kD;
    for (int d = dbase; d < dbase + 32; d += 4) {
      float4 a4 = *reinterpret_cast<const float4*>(r0 + d);
      float4 b4 = *reinterpret_cast<const float4*>(r1 + d);
      const float* af = reinterpret_cast<const float*>(&a4);
      const float* bf = reinterpret_cast<const float*>(&b4);
      for (int j = 0; j < 4; ++j) {
        unsigned pw = packbf(af[j], bf[j]);
        *reinterpret_cast<unsigned*>(&bufT[(d + j) * CH + mp]) = pw;
      }
    }
    for (int j = 0; j < 4; ++j) S32[tid + NTHREADS * j] = 0.f;
    if (tid < CH) lrinv[tid] = rinv[b * kN + mc + tid];
    if (mc + CH < kN)
      __builtin_prefetch(src + (size_t)CH * kD + tid * 64, 0, 1);  // next chunk
    __syncthreads();

    // --- GEMM1: S(16x32) += Anorm(16 x Kslice) * chunk^T, K split by wave ---
    {
      v8f c0 = {}, c1 = {};
      for (int t = 0; t < 4; ++t) {
        int k0 = q * 128 + t * 32;
        v16bf b0, b1;
        for (int s = 0; s < 16; ++s) {
          int kk = k0 + ((s < 8) ? shalf + s : 16 + shalf + (s - 8));
          b0[s] = bufT[kk * CH + (lane & 15)];
          b1[s] = bufT[kk * CH + 16 + (lane & 15)];
        }
        c0 = __builtin_amdgcn_wmma_f32_16x16x32_bf16(false, afrag[t], false, b0,
                                                     (short)0, c0, false, false);
        c1 = __builtin_amdgcn_wmma_f32_16x16x32_bf16(false, afrag[t], false, b1,
                                                     (short)0, c1, false, false);
      }
      int cn = lane & 15;
      for (int v = 0; v < 8; ++v) {
        int M = rt * 16 + ((lane < 16) ? v : v + 8);   // C-layout row mapping
        atomicAdd(&S32[M * CH + cn],      c0[v]);      // ds_add_f32 reduction
        atomicAdd(&S32[M * CH + 16 + cn], c1[v]);
      }
    }
    __syncthreads();

    // --- GEMM2: acc += tanh(S * rinv_m) * chunk ---
    {
      // vectorized S/rinv reads: 8x ds_load_b128 instead of 32x ds_load_b32
      const float4* sp4 = reinterpret_cast<const float4*>(
          &S32[(rt * 16 + (lane & 15)) * CH]);
      const float4* lr4 = reinterpret_cast<const float4*>(&lrinv[0]);
      const int o = shalf >> 2;  // 0 or 2
      float4 s0 = sp4[o], s1 = sp4[o + 1], s2 = sp4[o + 4], s3 = sp4[o + 5];
      float4 l0 = lr4[o], l1 = lr4[o + 1], l2 = lr4[o + 4], l3 = lr4[o + 5];
      const float* sf = reinterpret_cast<const float*>(&s0);  // s0..s3 contiguous
      const float* lf = reinterpret_cast<const float*>(&l0);
      float tv[16];
      for (int j = 0; j < 16; ++j) tv[j] = htanh(sf[j] * lf[j]);
      v8u asw;
      for (int j = 0; j < 8; ++j) asw[j] = packbf(tv[2 * j], tv[2 * j + 1]);
      v16bf as = __builtin_bit_cast(v16bf, asw);

      for (int i = 0; i < 8; ++i) {
        int d0 = q * 128 + i * 16;
        // B tile (32m x 16d) from transposed bufT via plain LDS loads:
        // lane reads row d, m = 8h..8h+7 and 16+8h..16+8h+7 + dscnt wait,
        // fused in one asm so the wait cannot be scheduled past the consumer.
        unsigned base = (unsigned)(size_t)&bufT[d0 * CH] + trlane;
        v4u blo, bhi;
        asm volatile("ds_load_b128 %0, %2\n\t"
                     "ds_load_b128 %1, %2 offset:32\n\t"
                     "s_wait_dscnt 0x0"
                     : "=&v"(blo), "=&v"(bhi) : "v"(base) : "memory");
        v8u   bbu = __builtin_shufflevector(blo, bhi, 0, 1, 2, 3, 4, 5, 6, 7);
        v16bf bb  = __builtin_bit_cast(v16bf, bbu);
        acc[i] = __builtin_amdgcn_wmma_f32_16x16x32_bf16(false, as, false, bb,
                                                         (short)0, acc[i], false, false);
      }
    }
  }

  // --- write this wave's (16 rows x 128 cols) slice of out ---
  for (int i = 0; i < 8; ++i) {
    int d0 = q * 128 + i * 16;
    for (int v = 0; v < 8; ++v) {
      int r = rt * 16 + ((lane < 16) ? v : v + 8);
      out[(size_t)(b * kN + blockRow + r) * kD + d0 + (lane & 15)] = acc[i][v];
    }
  }
}

extern "C" void kernel_launch(void* const* d_in, const int* in_sizes, int n_in,
                              void* d_out, int out_size, void* d_ws, size_t ws_size,
                              hipStream_t stream) {
  (void)in_sizes; (void)n_in; (void)out_size; (void)ws_size;
  const float* tgt = (const float*)d_in[0];   // query_pos (d_in[1]) unused by the math
  float* out  = (float*)d_out;
  float* rinv = (float*)d_ws;                 // kB*kN floats = 128 KB scratch

  age_rownorm<<<(kB * kN) / 8, 256, 0, stream>>>(tgt, rinv);
  age_fused<<<kB * (kN / ROWS), NTHREADS, 0, stream>>>(tgt, rinv, out);
}
